// Net_84817014162238
// MI455X (gfx1250) — matrix operand order repaired and checked by reading, not compiled, not run on hardware
//
#include <hip/hip_runtime.h>
#include <hip/hip_bf16.h>
#include <math.h>

typedef __attribute__((ext_vector_type(16))) _Float16 v16h;
typedef __attribute__((ext_vector_type(8)))  float    v8f;

#define BN_EPS 1e-5f
#define NEG_INF (-__builtin_inff())

union FragU { v16h h; unsigned u[8]; };

// A fragment (16x32 f16, row-major source, ld = row stride in halves, multiple of 32).
// ISA layout: lanes 0-15 row M=lane, VGPR0..3 = K 0..7, VGPR4..7 = K16..23; hi lanes +8 in K.
__device__ __forceinline__ v16h load_a_frag(const _Float16* tile, int ld, int lane) {
  int row = lane & 15;
  int hi4 = ((lane >> 4) & 1) * 4;
  const unsigned* p = (const unsigned*)(tile + (size_t)row * ld);
  FragU f;
#pragma unroll
  for (int v = 0; v < 4; ++v) { f.u[v] = p[v + hi4]; f.u[v + 4] = p[v + 8 + hi4]; }
  return f.h;
}

// B fragment (32x16 f16) from weights stored transposed [N][Kpad]:
// lane holds col n = lane&15; same K order as A: VGPR0..3 = K0..7, VGPR4..7 = K16..23; hi lanes +8 in K.
// colptr = WT + col*Kpad + kstep*32  (caller adds col offset).
__device__ __forceinline__ v16h load_b_frag(const _Float16* colptr, int lane) {
  int hi4 = ((lane >> 4) & 1) * 4;
  const unsigned* p = (const unsigned*)colptr;
  FragU f;
#pragma unroll
  for (int v = 0; v < 4; ++v) { f.u[v] = p[v + hi4]; f.u[v + 4] = p[v + 8 + hi4]; }
  return f.h;
}

// ---------------- weight prep: f32 [K][N] -> f16 transposed [N][Kpad] (zero pad) -----------
__global__ void wprep_kernel(const float* __restrict__ W, _Float16* __restrict__ WT,
                             int K, int N, int KP) {
  int t = blockIdx.x * blockDim.x + threadIdx.x;
  if (t >= N * KP) return;
  int n = t / KP, k = t - n * KP;
  WT[t] = (_Float16)(k < K ? W[(size_t)k * N + n] : 0.0f);
}

// ---------------- farthest point sampling: one block per batch ------------------------------
__global__ __launch_bounds__(256) void fps_kernel(const float* __restrict__ pos,
                                                  int n, int m, int* __restrict__ idx_out) {
  __shared__ float dist[4096];
  __shared__ float rv[256];
  __shared__ int ri[256];
  __shared__ int lastS;
  int b = blockIdx.x, tid = threadIdx.x;
  const float* pb = pos + (size_t)b * n * 3;
  for (int i = tid; i < n; i += 256) dist[i] = 3.4e38f;
  if (tid == 0) { lastS = 0; idx_out[(size_t)b * m] = 0; }
  __syncthreads();
  for (int s = 1; s < m; ++s) {
    int last = lastS;
    float lx = pb[last * 3 + 0], ly = pb[last * 3 + 1], lz = pb[last * 3 + 2];
    float best = -1.0f; int bi = 0x7fffffff;
    for (int i = tid; i < n; i += 256) {
      float dx = pb[i * 3 + 0] - lx, dy = pb[i * 3 + 1] - ly, dz = pb[i * 3 + 2] - lz;
      float d = dx * dx + dy * dy + dz * dz;
      float dm = fminf(dist[i], d);
      dist[i] = dm;
      if (dm > best) { best = dm; bi = i; }
    }
    rv[tid] = best; ri[tid] = bi;
    __syncthreads();
    for (int off = 128; off > 0; off >>= 1) {
      if (tid < off) {
        float ov = rv[tid + off]; int oi = ri[tid + off];
        if (ov > rv[tid] || (ov == rv[tid] && oi < ri[tid])) { rv[tid] = ov; ri[tid] = oi; }
      }
      __syncthreads();
    }
    if (tid == 0) { lastS = ri[0]; idx_out[(size_t)b * m + s] = ri[0]; }
    __syncthreads();
  }
}

// ---------------- gather sampled positions --------------------------------------------------
__global__ void gather_pos_kernel(const float* __restrict__ pos, const int* __restrict__ idx,
                                  float* __restrict__ qpos, int n, int m, int total) {
  int t = blockIdx.x * blockDim.x + threadIdx.x;
  if (t >= total) return;
  int b = t / m;
  int s = idx[t];
  qpos[(size_t)t * 3 + 0] = pos[((size_t)b * n + s) * 3 + 0];
  qpos[(size_t)t * 3 + 1] = pos[((size_t)b * n + s) * 3 + 1];
  qpos[(size_t)t * 3 + 2] = pos[((size_t)b * n + s) * 3 + 2];
}

// ---------------- ball query (top-64 nearest, radius mask): one block per query -------------
__global__ __launch_bounds__(256) void ballq_kernel(const float* __restrict__ spos,
                                                    const float* __restrict__ qpos,
                                                    int n, int m, float r2,
                                                    int* __restrict__ nidx,
                                                    int* __restrict__ nmask) {
  __shared__ float d2[4096];
  __shared__ float rv[256];
  __shared__ int ri[256];
  int q = blockIdx.x, tid = threadIdx.x;
  int b = q / m;
  float qx = qpos[(size_t)q * 3 + 0], qy = qpos[(size_t)q * 3 + 1], qz = qpos[(size_t)q * 3 + 2];
  const float* sp = spos + (size_t)b * n * 3;
  for (int i = tid; i < n; i += 256) {
    float dx = sp[i * 3 + 0] - qx, dy = sp[i * 3 + 1] - qy, dz = sp[i * 3 + 2] - qz;
    d2[i] = dx * dx + dy * dy + dz * dz;
  }
  __syncthreads();
  for (int k = 0; k < 64; ++k) {
    float best = 3.5e38f; int bi = 0x7fffffff;
    for (int i = tid; i < n; i += 256) {
      float v = d2[i];
      if (v < best) { best = v; bi = i; }
    }
    rv[tid] = best; ri[tid] = bi;
    __syncthreads();
    for (int off = 128; off > 0; off >>= 1) {
      if (tid < off) {
        if (rv[tid + off] < rv[tid] || (rv[tid + off] == rv[tid] && ri[tid + off] < ri[tid])) {
          rv[tid] = rv[tid + off]; ri[tid] = ri[tid + off];
        }
      }
      __syncthreads();
    }
    if (tid == 0) {
      int sel = ri[0];
      nidx[(size_t)q * 64 + k] = sel;
      nmask[(size_t)q * 64 + k] = (rv[0] <= r2) ? 1 : 0;
      d2[sel] = 3.5e38f;
    }
    __syncthreads();
  }
}

// ---------------- fused PointConv (2-layer MLP + masked max agg) via WMMA -------------------
// One workgroup (4 wave32) per query; wave w owns 16-neighbor row tile w.
template <int CF, int CINP, int CH, int COUT>
__global__ __launch_bounds__(128)
void pconv_kernel(const float* __restrict__ x,     // [B][n][CF]
                  const float* __restrict__ spos,  // [B][n][3]
                  const float* __restrict__ qpos,  // [B][m][3]
                  const int* __restrict__ nidx,    // [B*m][64]
                  const int* __restrict__ nmask,   // [B*m][64]
                  const _Float16* __restrict__ W1T, const float* __restrict__ b1,
                  const float* __restrict__ g1, const float* __restrict__ be1,
                  const _Float16* __restrict__ W2T, const float* __restrict__ b2,
                  const float* __restrict__ g2, const float* __restrict__ be2,
                  float* __restrict__ out, int outStride, int outOff,
                  int n, int m) {
  constexpr int CIN = CF + 3;
  __shared__ alignas(16) _Float16 Xs[64 * CINP];
  __shared__ alignas(16) _Float16 Hs[64 * CH];
  __shared__ float outPart[4][COUT];
  int q = blockIdx.x;
  int b = q / m, qi = q - b * m;
  int tid = threadIdx.x;
  const int* qidx = nidx + (size_t)q * 64;
  const int* qmsk = nmask + (size_t)q * 64;
  float qp0 = qpos[((size_t)b * m + qi) * 3 + 0];
  float qp1 = qpos[((size_t)b * m + qi) * 3 + 1];
  float qp2 = qpos[((size_t)b * m + qi) * 3 + 2];

  // stage gathered features (x_j, pos_j - pos_i) as f16, zero-padded to CINP
  for (int e = tid; e < 64 * CINP; e += 128) {
    int j = e / CINP, c = e - j * CINP;
    int nb = qidx[j];
    float v = 0.0f;
    if (c < CF) {
      v = x[((size_t)b * n + nb) * (size_t)CF + c];
    } else if (c < CIN) {
      int d = c - CF;
      float pv = spos[((size_t)b * n + nb) * 3 + d];
      v = pv - (d == 0 ? qp0 : (d == 1 ? qp1 : qp2));
    }
    Xs[e] = (_Float16)v;
  }
  __syncthreads();

  int wv = tid >> 5, lane = tid & 31;
  int nlo = lane & 15, hi = (lane >> 4) & 1;
  const float bnS = rsqrtf(1.0f + BN_EPS);

  // layer 1: X[64,CINP] @ W1[CINP,CH] -> ReLU -> BN -> Hs (f16)
  for (int ct = 0; ct < CH / 16; ++ct) {
    v8f acc = {};
#pragma unroll
    for (int ks = 0; ks < CINP / 32; ++ks) {
      v16h a = load_a_frag(Xs + (size_t)wv * 16 * CINP + ks * 32, CINP, lane);
      v16h bf = load_b_frag(W1T + (size_t)(ct * 16 + nlo) * CINP + ks * 32, lane);
      acc = __builtin_amdgcn_wmma_f32_16x16x32_f16(false, a, false, bf, (short)0, acc, false, false);
    }
    int col = ct * 16 + nlo;
    float bb = b1[col], sc = g1[col] * bnS, sh = be1[col];
#pragma unroll
    for (int r = 0; r < 8; ++r) {
      float v = acc[r] + bb;
      v = v > 0.0f ? v : 0.0f;
      v = v * sc + sh;
      int row = wv * 16 + hi * 8 + r;
      Hs[(size_t)row * CH + col] = (_Float16)v;
    }
  }
  __syncthreads();

  // layer 2: Hs[64,CH] @ W2[CH,COUT] -> ReLU -> BN -> mask -> max over neighbors
  for (int ct = 0; ct < COUT / 16; ++ct) {
    v8f acc = {};
#pragma unroll
    for (int ks = 0; ks < CH / 32; ++ks) {
      v16h a = load_a_frag(Hs + (size_t)wv * 16 * CH + ks * 32, CH, lane);
      v16h bf = load_b_frag(W2T + (size_t)(ct * 16 + nlo) * CH + ks * 32, lane);
      acc = __builtin_amdgcn_wmma_f32_16x16x32_f16(false, a, false, bf, (short)0, acc, false, false);
    }
    int col = ct * 16 + nlo;
    float bb = b2[col], sc = g2[col] * bnS, sh = be2[col];
    float pm = NEG_INF;
#pragma unroll
    for (int r = 0; r < 8; ++r) {
      float v = acc[r] + bb;
      v = v > 0.0f ? v : 0.0f;
      v = v * sc + sh;
      int row = wv * 16 + hi * 8 + r;
      v = qmsk[row] ? v : NEG_INF;
      pm = fmaxf(pm, v);
    }
    pm = fmaxf(pm, __shfl_xor(pm, 16, 32));
    if (hi == 0) outPart[wv][col] = pm;
  }
  __syncthreads();

  for (int t = tid; t < COUT; t += 128) {
    float mv = fmaxf(fmaxf(outPart[0][t], outPart[1][t]), fmaxf(outPart[2][t], outPart[3][t]));
    if (!(mv > NEG_INF)) mv = 0.0f;  // no neighbor inside radius -> 0
    out[(size_t)q * outStride + outOff + t] = mv;
  }
}

// ---------------- p3 MLP (515 -> 1024) via WMMA, 32-row tiles -------------------------------
__global__ __launch_bounds__(64)
void mlp3_kernel(const float* __restrict__ x2,    // [B][M2][512]
                 const float* __restrict__ qpos2, // [B][M2][3]
                 const _Float16* __restrict__ W3T, // [1024][544]
                 const float* __restrict__ b3, const float* __restrict__ g3,
                 const float* __restrict__ be3,
                 float* __restrict__ h3,          // [B][416][1024]
                 int M2) {
  constexpr int KP = 544, NC = 1024, TPB = 13;
  __shared__ alignas(16) _Float16 Xs[32 * KP];
  int blk = blockIdx.x;
  int b = blk / TPB, rt0 = (blk - b * TPB) * 32;
  int tid = threadIdx.x;
  for (int e = tid; e < 32 * KP; e += 64) {
    int j = e / KP, c = e - j * KP;
    int row = rt0 + j;
    float v = 0.0f;
    if (row < M2) {
      if (c < 512) v = x2[((size_t)b * M2 + row) * 512 + c];
      else if (c < 515) v = qpos2[((size_t)b * M2 + row) * 3 + (c - 512)];
    }
    Xs[e] = (_Float16)v;
  }
  __syncthreads();
  int wv = tid >> 5, lane = tid & 31;
  int nlo = lane & 15, hi = (lane >> 4) & 1;
  const float bnS = rsqrtf(1.0f + BN_EPS);
  for (int ct = 0; ct < NC / 16; ++ct) {
    v8f acc = {};
#pragma unroll
    for (int ks = 0; ks < KP / 32; ++ks) {
      v16h a = load_a_frag(Xs + (size_t)wv * 16 * KP + ks * 32, KP, lane);
      v16h bf = load_b_frag(W3T + (size_t)(ct * 16 + nlo) * KP + ks * 32, lane);
      acc = __builtin_amdgcn_wmma_f32_16x16x32_f16(false, a, false, bf, (short)0, acc, false, false);
    }
    int col = ct * 16 + nlo;
    float bb = b3[col], sc = g3[col] * bnS, sh = be3[col];
#pragma unroll
    for (int r = 0; r < 8; ++r) {
      float v = acc[r] + bb;
      v = v > 0.0f ? v : 0.0f;
      v = v * sc + sh;
      int row = rt0 + wv * 16 + hi * 8 + r;
      h3[((size_t)b * 416 + row) * 1024 + col] = v;
    }
  }
}

// ---------------- global max pool over M2 rows ---------------------------------------------
__global__ void maxreduce_kernel(const float* __restrict__ h3, float* __restrict__ g,
                                 int M2, int rowsPad, int C, int total) {
  int t = blockIdx.x * blockDim.x + threadIdx.x;
  if (t >= total) return;
  int b = t / C, c = t - b * C;
  float m = NEG_INF;
  for (int r = 0; r < M2; ++r) m = fmaxf(m, h3[((size_t)b * rowsPad + r) * C + c]);
  g[t] = m;
}

// ---------------- FC head -------------------------------------------------------------------
__global__ __launch_bounds__(512) void fc1_kernel(const float* __restrict__ g,
                                                  const float* __restrict__ wl1,
                                                  const float* __restrict__ bl1,
                                                  float* __restrict__ h1) {
  int b = blockIdx.x, col = threadIdx.x;
  float s = bl1[col];
  const float* gb = g + (size_t)b * 1024;
  for (int k = 0; k < 1024; ++k) s += gb[k] * wl1[(size_t)k * 512 + col];
  h1[(size_t)b * 512 + col] = s > 0.0f ? s : 0.0f;
}

__global__ __launch_bounds__(128) void fc2norm_kernel(const float* __restrict__ h1,
                                                      const float* __restrict__ wl2,
                                                      const float* __restrict__ bl2,
                                                      float* __restrict__ out) {
  __shared__ float ss[128];
  int b = blockIdx.x, t = threadIdx.x;
  float o = bl2[t];
  const float* hb = h1 + (size_t)b * 512;
  for (int k = 0; k < 512; ++k) o += hb[k] * wl2[(size_t)k * 128 + t];
  ss[t] = o * o;
  __syncthreads();
  for (int off = 64; off > 0; off >>= 1) {
    if (t < off) ss[t] += ss[t + off];
    __syncthreads();
  }
  out[(size_t)b * 128 + t] = o / sqrtf(ss[0]);
}

// ---------------- launch --------------------------------------------------------------------
extern "C" void kernel_launch(void* const* d_in, const int* in_sizes, int n_in,
                              void* d_out, int out_size, void* d_ws, size_t ws_size,
                              hipStream_t stream) {
  (void)in_sizes; (void)n_in; (void)out_size; (void)ws_size;
  const int B = 4, N = 4096, M1 = 2048, M2 = 410;

  const float* xyz   = (const float*)d_in[0];
  const float* color = (const float*)d_in[1];
  const float* w1a = (const float*)d_in[2];  const float* b1a = (const float*)d_in[3];
  const float* g1a = (const float*)d_in[4];  const float* be1a = (const float*)d_in[5];
  const float* w1b = (const float*)d_in[6];  const float* b1b = (const float*)d_in[7];
  const float* g1b = (const float*)d_in[8];  const float* be1b = (const float*)d_in[9];
  const float* w2a = (const float*)d_in[10]; const float* b2a = (const float*)d_in[11];
  const float* g2a = (const float*)d_in[12]; const float* be2a = (const float*)d_in[13];
  const float* w2b = (const float*)d_in[14]; const float* b2b = (const float*)d_in[15];
  const float* g2b = (const float*)d_in[16]; const float* be2b = (const float*)d_in[17];
  const float* w3  = (const float*)d_in[18]; const float* b3 = (const float*)d_in[19];
  const float* g3  = (const float*)d_in[20]; const float* be3 = (const float*)d_in[21];
  const float* wl1 = (const float*)d_in[22]; const float* bl1 = (const float*)d_in[23];
  const float* wl2 = (const float*)d_in[24]; const float* bl2 = (const float*)d_in[25];

  char* ws = (char*)d_ws;
  size_t off = 0;
  auto take = [&](size_t bytes) -> char* {
    off = (off + 255) & ~(size_t)255;
    char* p = ws + off;
    off += bytes;
    return p;
  };
  _Float16* W1aT = (_Float16*)take((size_t)64 * 32 * 2);
  _Float16* W1bT = (_Float16*)take((size_t)64 * 64 * 2);
  _Float16* W2aT = (_Float16*)take((size_t)128 * 160 * 2);
  _Float16* W2bT = (_Float16*)take((size_t)256 * 128 * 2);
  _Float16* W3T  = (_Float16*)take((size_t)1024 * 544 * 2);
  int*   idx1  = (int*)take((size_t)B * M1 * 4);
  float* qpos1 = (float*)take((size_t)B * M1 * 3 * 4);
  int*   nidx  = (int*)take((size_t)B * M1 * 64 * 4);
  int*   nmask = (int*)take((size_t)B * M1 * 64 * 4);
  float* x1    = (float*)take((size_t)B * M1 * 128 * 4);
  int*   idx2  = (int*)take((size_t)B * M2 * 4);
  float* qpos2 = (float*)take((size_t)B * M2 * 3 * 4);
  float* x2    = (float*)take((size_t)B * M2 * 512 * 4);
  float* h3    = (float*)take((size_t)B * 416 * 1024 * 4);
  float* gbuf  = (float*)take((size_t)B * 1024 * 4);
  float* h1    = (float*)take((size_t)B * 512 * 4);

  // weight prep (f32 -> f16 transposed, K padded)
  wprep_kernel<<<(64 * 32 + 255) / 256, 256, 0, stream>>>(w1a, W1aT, 6, 64, 32);
  wprep_kernel<<<(64 * 64 + 255) / 256, 256, 0, stream>>>(w1b, W1bT, 64, 64, 64);
  wprep_kernel<<<(128 * 160 + 255) / 256, 256, 0, stream>>>(w2a, W2aT, 131, 128, 160);
  wprep_kernel<<<(256 * 128 + 255) / 256, 256, 0, stream>>>(w2b, W2bT, 128, 256, 128);
  wprep_kernel<<<(1024 * 544 + 255) / 256, 256, 0, stream>>>(w3, W3T, 515, 1024, 544);

  // SA module 1
  fps_kernel<<<B, 256, 0, stream>>>(xyz, N, M1, idx1);
  gather_pos_kernel<<<(B * M1 + 255) / 256, 256, 0, stream>>>(xyz, idx1, qpos1, N, M1, B * M1);
  const float rad1[2] = {0.2f, 0.1f};
  for (int ri = 0; ri < 2; ++ri) {
    ballq_kernel<<<B * M1, 256, 0, stream>>>(xyz, qpos1, N, M1, rad1[ri] * rad1[ri], nidx, nmask);
    pconv_kernel<3, 32, 64, 64><<<B * M1, 128, 0, stream>>>(
        color, xyz, qpos1, nidx, nmask,
        W1aT, b1a, g1a, be1a, W1bT, b1b, g1b, be1b,
        x1, 128, ri * 64, N, M1);
  }

  // SA module 2
  fps_kernel<<<B, 256, 0, stream>>>(qpos1, M1, M2, idx2);
  gather_pos_kernel<<<(B * M2 + 255) / 256, 256, 0, stream>>>(qpos1, idx2, qpos2, M1, M2, B * M2);
  const float rad2[2] = {0.35f, 0.5f};
  for (int ri = 0; ri < 2; ++ri) {
    ballq_kernel<<<B * M2, 256, 0, stream>>>(qpos1, qpos2, M1, M2, rad2[ri] * rad2[ri], nidx, nmask);
    pconv_kernel<128, 160, 128, 256><<<B * M2, 128, 0, stream>>>(
        x1, qpos1, qpos2, nidx, nmask,
        W2aT, b2a, g2a, be2a, W2bT, b2b, g2b, be2b,
        x2, 512, ri * 256, M1, M2);
  }

  // global MLP + max pool + FC head
  mlp3_kernel<<<B * 13, 64, 0, stream>>>(x2, qpos2, W3T, b3, g3, be3, h3, M2);
  maxreduce_kernel<<<(B * 1024 + 255) / 256, 256, 0, stream>>>(h3, gbuf, M2, 416, 1024, B * 1024);
  fc1_kernel<<<B, 512, 0, stream>>>(gbuf, wl1, bl1, h1);
  fc2norm_kernel<<<B, 128, 0, stream>>>(h1, wl2, bl2, (float*)d_out);
}
